// LayerG2_88038239633789
// MI455X (gfx1250) — matrix operand order repaired and checked by reading, not compiled, not run on hardware
//
#include <hip/hip_runtime.h>

typedef __attribute__((ext_vector_type(16))) __bf16 v16bf;
typedef __attribute__((ext_vector_type(8)))  __bf16 v8bf;
typedef __attribute__((ext_vector_type(8)))  float  v8f;

#define NNODES 100000
#define NEDGES 1600000
#define NIN    512
#define NOUT   64
#define VAR_EPS 1e-4f

// ---------------------------------------------------------------------------
// Convert weights to bf16; gcn_weight [512,64] (k-major) -> bg [64,512]
// (n-major, transposed) so B fragments are contiguous; var_weight [64,512]
// is already n-major.
// ---------------------------------------------------------------------------
__global__ __launch_bounds__(256) void prep_weights_kernel(
    const float* __restrict__ gw, const float* __restrict__ vw,
    __bf16* __restrict__ bg, __bf16* __restrict__ bv) {
  int tid = blockIdx.x * blockDim.x + threadIdx.x;
  if (tid < NOUT * NIN) {
    int n = tid >> 9;          // NIN == 512
    int k = tid & (NIN - 1);
    bg[tid] = (__bf16)gw[k * NOUT + n];
    bv[tid] = (__bf16)vw[tid];
  }
}

__global__ __launch_bounds__(256) void zero_kernel(float* __restrict__ p, int n) {
  int t = blockIdx.x * blockDim.x + threadIdx.x;
  if (t < n) p[t] = 0.0f;
}

// ---------------------------------------------------------------------------
// Fused dual GEMM: one wave computes a 16-row x 64-col strip of both
// support = x @ Wgcn and logits = x @ Wvar^T using v_wmma_f32_16x16x32_bf16.
// Epilogue: support -> workspace, std = sqrt(exp(logit + bias) + eps) -> out.
// ---------------------------------------------------------------------------
__global__ __launch_bounds__(256) void gemm_fused_wmma_kernel(
    const float* __restrict__ x,
    const __bf16* __restrict__ bg,   // [64][512]  gcn^T, bf16
    const __bf16* __restrict__ bv,   // [64][512]  var,   bf16
    const float* __restrict__ bias,  // [64]
    float* __restrict__ support,     // [N][64]
    float* __restrict__ stdv) {      // [N][64]
  const int lane = threadIdx.x & 31;
  const int wave = threadIdx.x >> 5;
  const int rowBase = blockIdx.x * 128 + wave * 16;
  const int m  = lane & 15;   // A row-in-tile, B/C column-in-tile
  const int kh = lane >> 4;   // K-half selector

  int rA = rowBase + m;
  if (rA >= NNODES) rA = NNODES - 1;          // clamp OOB reads; stores guarded
  const float* xrow = x + (size_t)rA * NIN;

  v8f zero = {};
  v8f accg[4], accv[4];
#pragma unroll
  for (int i = 0; i < 4; ++i) { accg[i] = zero; accv[i] = zero; }

  for (int kk = 0; kk < NIN; kk += 32) {
    // --- A fragment: elements 0..15 = K(kk + kh*16 + 0..15)
    const float4 fa = *(const float4*)(xrow + kk + kh * 16);
    const float4 fb = *(const float4*)(xrow + kk + kh * 16 + 4);
    const float4 fc = *(const float4*)(xrow + kk + kh * 16 + 8);
    const float4 fd = *(const float4*)(xrow + kk + kh * 16 + 12);
    v16bf a;
    a[0]  = (__bf16)fa.x; a[1]  = (__bf16)fa.y; a[2]  = (__bf16)fa.z; a[3]  = (__bf16)fa.w;
    a[4]  = (__bf16)fb.x; a[5]  = (__bf16)fb.y; a[6]  = (__bf16)fb.z; a[7]  = (__bf16)fb.w;
    a[8]  = (__bf16)fc.x; a[9]  = (__bf16)fc.y; a[10] = (__bf16)fc.z; a[11] = (__bf16)fc.w;
    a[12] = (__bf16)fd.x; a[13] = (__bf16)fd.y; a[14] = (__bf16)fd.z; a[15] = (__bf16)fd.w;

#pragma unroll
    for (int nt = 0; nt < 4; ++nt) {
      const int col = nt * 16 + m;
      // B fragment: lane holds column `col`, 16 consecutive K from kk + kh*16
      const __bf16* pg = bg + (size_t)col * NIN + kk + kh * 16;
      const __bf16* pv = bv + (size_t)col * NIN + kk + kh * 16;
      v8bf glo = *(const v8bf*)(pg);
      v8bf ghi = *(const v8bf*)(pg + 8);
      v8bf vlo = *(const v8bf*)(pv);
      v8bf vhi = *(const v8bf*)(pv + 8);
      v16bf bG = __builtin_shufflevector(glo, ghi, 0, 1, 2, 3, 4, 5, 6, 7,
                                         8, 9, 10, 11, 12, 13, 14, 15);
      v16bf bV = __builtin_shufflevector(vlo, vhi, 0, 1, 2, 3, 4, 5, 6, 7,
                                         8, 9, 10, 11, 12, 13, 14, 15);
      accg[nt] = __builtin_amdgcn_wmma_f32_16x16x32_bf16(
          false, a, false, bG, (short)0, accg[nt], false, false);
      accv[nt] = __builtin_amdgcn_wmma_f32_16x16x32_bf16(
          false, a, false, bV, (short)0, accv[nt], false, false);
    }
  }

  // --- epilogue: C/D layout -> row = rowBase + kh*8 + v, col = nt*16 + m
#pragma unroll
  for (int nt = 0; nt < 4; ++nt) {
    const int col = nt * 16 + m;
    const float b = bias[col];
#pragma unroll
    for (int v = 0; v < 8; ++v) {
      int row = rowBase + kh * 8 + v;
      if (row < NNODES) {
        support[(size_t)row * NOUT + col] = accg[nt][v];
        float qv = __expf(accv[nt][v] + b) + VAR_EPS;
        stdv[(size_t)row * NOUT + col] = __builtin_sqrtf(qv);
      }
    }
  }
}

// ---------------------------------------------------------------------------
// Edge scatter: q_m[dst] += w * support[src]; thread per (edge, 4-feature chunk)
// ---------------------------------------------------------------------------
__global__ __launch_bounds__(256) void edge_scatter_kernel(
    const int* __restrict__ esrc, const int* __restrict__ edst,
    const float* __restrict__ ew, const float* __restrict__ support,
    float* __restrict__ qm) {
  long tid = (long)blockIdx.x * blockDim.x + threadIdx.x;
  if (tid >= (long)NEDGES * 16) return;
  int e = (int)(tid >> 4);
  int c = ((int)tid & 15) << 2;
  int s = esrc[e];
  int d = edst[e];
  float w = ew[e];
  const float4 sv = *(const float4*)(support + (size_t)s * NOUT + c);
  float* q = qm + (size_t)d * NOUT + c;
  unsafeAtomicAdd(q + 0, w * sv.x);
  unsafeAtomicAdd(q + 1, w * sv.y);
  unsafeAtomicAdd(q + 2, w * sv.z);
  unsafeAtomicAdd(q + 3, w * sv.w);
}

__global__ __launch_bounds__(256) void finalize_kernel(
    const float* __restrict__ qm, const float* __restrict__ stdv,
    const float* __restrict__ noise, float* __restrict__ latent, int n) {
  int t = blockIdx.x * blockDim.x + threadIdx.x;
  if (t < n) latent[t] = qm[t] + stdv[t] * noise[t];
}

// ---------------------------------------------------------------------------
extern "C" void kernel_launch(void* const* d_in, const int* in_sizes, int n_in,
                              void* d_out, int out_size, void* d_ws, size_t ws_size,
                              hipStream_t stream) {
  const float* x     = (const float*)d_in[0];
  const int*   esrc  = (const int*)d_in[1];
  const int*   edst  = (const int*)d_in[2];
  const float* ew    = (const float*)d_in[3];
  const float* noise = (const float*)d_in[4];
  const float* gw    = (const float*)d_in[5];
  const float* vw    = (const float*)d_in[6];
  const float* vb    = (const float*)d_in[7];

  float* out    = (float*)d_out;
  float* qm     = out;                                  // [N,64]
  float* stdv   = out + (size_t)NNODES * NOUT;          // [N,64]
  float* latent = stdv + (size_t)NNODES * NOUT;         // [N,64]

  char*   ws      = (char*)d_ws;
  float*  support = (float*)ws;                                        // 25.6 MB
  __bf16* bg      = (__bf16*)(ws + (size_t)NNODES * NOUT * sizeof(float));
  __bf16* bv      = bg + (size_t)NOUT * NIN;

  int nElems = NNODES * NOUT;

  prep_weights_kernel<<<(NOUT * NIN + 255) / 256, 256, 0, stream>>>(gw, vw, bg, bv);
  zero_kernel<<<(nElems + 255) / 256, 256, 0, stream>>>(qm, nElems);
  gemm_fused_wmma_kernel<<<(NNODES + 127) / 128, 256, 0, stream>>>(
      x, bg, bv, vb, support, stdv);
  edge_scatter_kernel<<<(int)(((long)NEDGES * 16 + 255) / 256), 256, 0, stream>>>(
      esrc, edst, ew, support, qm);
  finalize_kernel<<<(nElems + 255) / 256, 256, 0, stream>>>(
      qm, stdv, noise, latent, nElems);
}
